// LstmRnn_39024072852069
// MI455X (gfx1250) — matrix operand order repaired and checked by reading, not compiled, not run on hardware
//
#include <hip/hip_runtime.h>

// ---------------- CDNA5 / gfx1250 WMMA types ----------------
typedef __attribute__((ext_vector_type(16))) _Float16 v16h;
typedef __attribute__((ext_vector_type(8)))  _Float16 v8h;
typedef __attribute__((ext_vector_type(8)))  float    v8f;

#define Bdim 64
#define Ldim 1024
#define Idim 512
#define Hdim 512
#define G4   2048   // 4*H gate rows

__device__ __forceinline__ float sigmoidf_(float x) { return 1.0f / (1.0f + __expf(-x)); }
__device__ __forceinline__ float tanhfast_(float x) { return 2.0f / (1.0f + __expf(-2.0f * x)) - 1.0f; }

union AFrag { v16h v; v8h h[2]; };

// ---------------- prep: f32 -> f16 elementwise ----------------
__global__ void k_cvt_f16(const float* __restrict__ src, _Float16* __restrict__ dst, int n) {
    int i = blockIdx.x * blockDim.x + threadIdx.x;
    if (i < n) dst[i] = (_Float16)src[i];
}

// ---------------- prep: pack weight [2048, 512] row-major into WMMA B-fragment tile order
// Tile (nt, kt) covers gate rows nt*16..+15 (N) and k kt*32..+31 (K).
// B order used here (contiguous k, A loads match): N = lane&15, K = e + 16*(lane>>4), e = 0..15.
// Packed so each lane's 16 halves (32 B) are contiguous -> one aligned fragment load.
__global__ void k_pack_w(const float* __restrict__ w, _Float16* __restrict__ pk) {
    int id = blockIdx.x * blockDim.x + threadIdx.x;      // total = 2048*512
    if (id >= G4 * 512) return;
    int e    = id & 15;
    int lane = (id >> 4) & 31;
    int kt   = (id >> 9) & 15;        // 512/32 = 16 K-tiles
    int nt   = id >> 13;              // 2048/16 = 128 N-tiles
    int row  = nt * 16 + (lane & 15);
    int kk   = kt * 32 + e + ((lane >> 4) << 4);
    pk[id] = (_Float16)w[row * 512 + kk];
}

// ---------------- phase 1: x_gates = x @ w_ih^T + (b_ih + b_hh) ----------------
// GEMM [65536 x 2048] = [65536 x 512] x [512 x 2048], f16 WMMA, f32 accumulate.
// One wave = 2 M-tiles x 4 N-tiles (8 accumulators), K-loop fully unrolled with
// explicit double-buffered A/B fragments so loads pipeline ahead of the WMMAs.
__global__ __launch_bounds__(256) void k_xgates(const _Float16* __restrict__ xh,
                                                const _Float16* __restrict__ wpk,
                                                const float* __restrict__ bih,
                                                const float* __restrict__ bhh,
                                                float* __restrict__ xg) {
    const int lane = threadIdx.x & 31;
    const int wv   = blockIdx.x * (blockDim.x >> 5) + (threadIdx.x >> 5);
    const int mt0  = (wv >> 5) << 1;       // first of 2 M-tiles (0..4094)
    const int ntb  = (wv & 31) << 2;       // N-tile base: 0,4,...,124
    const int half = lane >> 4;
    const int lr   = lane & 15;

    v8f z = {};
    v8f acc[2][4];
#pragma unroll
    for (int m = 0; m < 2; ++m)
#pragma unroll
        for (int j = 0; j < 4; ++j) acc[m][j] = z;

    const _Float16* aptr0 = xh + (size_t)(mt0 * 16 + lr) * Idim;
    const _Float16* aptr1 = aptr0 + (size_t)16 * Idim;
    const _Float16* bcol  = wpk + (size_t)lane * 16;

    AFrag a0c, a1c, a0n, a1n;
    v16h  bc[4], bn[4];

    // prologue: kt = 0
    {
        int ak = half * 16;
        a0c.h[0] = *(const v8h*)(aptr0 + ak);
        a0c.h[1] = *(const v8h*)(aptr0 + ak + 8);
        a1c.h[0] = *(const v8h*)(aptr1 + ak);
        a1c.h[1] = *(const v8h*)(aptr1 + ak + 8);
#pragma unroll
        for (int j = 0; j < 4; ++j)
            bc[j] = *(const v16h*)(bcol + (size_t)(ntb + j) * 8192);
    }

#pragma unroll
    for (int kt = 0; kt < 16; ++kt) {
        if (kt < 15) {   // issue next iteration's loads before consuming current
            int ak = (kt + 1) * 32 + half * 16;
            a0n.h[0] = *(const v8h*)(aptr0 + ak);
            a0n.h[1] = *(const v8h*)(aptr0 + ak + 8);
            a1n.h[0] = *(const v8h*)(aptr1 + ak);
            a1n.h[1] = *(const v8h*)(aptr1 + ak + 8);
#pragma unroll
            for (int j = 0; j < 4; ++j)
                bn[j] = *(const v16h*)(bcol + (size_t)(kt + 1) * 512 + (size_t)(ntb + j) * 8192);
        }
#pragma unroll
        for (int j = 0; j < 4; ++j) {
            acc[0][j] = __builtin_amdgcn_wmma_f32_16x16x32_f16(false, a0c.v, false, bc[j], (short)0, acc[0][j], false, false);
            acc[1][j] = __builtin_amdgcn_wmma_f32_16x16x32_f16(false, a1c.v, false, bc[j], (short)0, acc[1][j], false, false);
        }
        if (kt < 15) {
            a0c = a0n; a1c = a1n;
#pragma unroll
            for (int j = 0; j < 4; ++j) bc[j] = bn[j];
        }
    }

#pragma unroll
    for (int m = 0; m < 2; ++m)
#pragma unroll
        for (int j = 0; j < 4; ++j) {
            int g = (ntb + j) * 16 + lr;
            float bias = bih[g] + bhh[g];
#pragma unroll
            for (int r = 0; r < 8; ++r) {
                int row = (mt0 + m) * 16 + r + half * 8;  // C/D layout: row = r + 8*(lane>=16)
                xg[(size_t)row * G4 + g] = acc[m][j][r] + bias;
            }
        }
}

// ---------------- phase 2: persistent recurrent scan ----------------
// 4 workgroups (one per 16-row batch tile; batch rows are recurrence-independent).
// 1024 threads = 32 waves = 8 waves/SIMD so the SIMD can round-robin across B-load
// waits. Each wave owns 1 j-tile (16 h-columns): 4 accumulators, 64 v_wmma/step,
// ~100 VGPRs -> fits the 128-VGPR budget at 8 waves/SIMD. Accumulators are seeded
// from x_gates (WMMA C-operand) so those loads hide under the GEMM. h/c state is
// register-resident; h mirrored to LDS (f16) for A fragments. One barrier pair/step.
__global__ __launch_bounds__(1024, 1) void k_lstm(const _Float16* __restrict__ wpk,
                                                  const float* __restrict__ xg,
                                                  const float* __restrict__ mask,
                                                  const float* __restrict__ h0,
                                                  const float* __restrict__ c0,
                                                  float* __restrict__ out) {
    __shared__ alignas(32) _Float16 hbuf[16 * Hdim];   // 16 KB of 320 KB WGP LDS

    const int lane = threadIdx.x & 31;
    const int jt   = threadIdx.x >> 5;      // wave id == j-tile, 0..31
    const int half = lane >> 4;
    const int lr   = lane & 15;
    const int bb0  = blockIdx.x * 16;       // batch tile base
    const int col  = jt * 16 + lr;          // this lane's h column (fixed forever)

    float c_reg[8], h_reg[8];
    {
        float hv = h0[col], cv = c0[col];
#pragma unroll
        for (int r = 0; r < 8; ++r) {
            h_reg[r] = hv;
            c_reg[r] = cv;
            hbuf[(r + half * 8) * Hdim + col] = (_Float16)hv;
        }
    }
    __syncthreads();

    const _Float16* wp = wpk + (size_t)lane * 16;

    for (int t = 0; t < Ldim; ++t) {
        // seed accumulators with x_gates (biases pre-folded): C operand of first WMMA.
        // 32 scalar loads issued up-front; latency hides under the K-loop below.
        v8f acc[4];
#pragma unroll
        for (int r = 0; r < 8; ++r) {
            size_t gb = ((size_t)(bb0 + r + half * 8) * Ldim + t) * (size_t)G4 + col;
            acc[0][r] = xg[gb];
            acc[1][r] = xg[gb + 512];
            acc[2][r] = xg[gb + 1024];
            acc[3][r] = xg[gb + 1536];
        }

        // gates += h @ w_hh^T : per kt, load all 4 B fragments first (one clause,
        // staggered waits), then run the 4 WMMAs.
#pragma unroll
        for (int kt = 0; kt < 16; ++kt) {
            AFrag a;
            int ak = kt * 32 + half * 16;
            a.h[0] = *(const v8h*)(&hbuf[lr * Hdim + ak]);
            a.h[1] = *(const v8h*)(&hbuf[lr * Hdim + ak + 8]);
            v16h b[4];
#pragma unroll
            for (int g = 0; g < 4; ++g)
                b[g] = *(const v16h*)(wp + (size_t)((g * 32 + jt) * 16 + kt) * 512);
#pragma unroll
            for (int g = 0; g < 4; ++g)
                acc[g] = __builtin_amdgcn_wmma_f32_16x16x32_f16(
                    false, a.v, false, b[g], (short)0, acc[g], false, false);
        }
        __syncthreads();   // all waves done reading hbuf

        // elementwise LSTM cell update (pure VALU); lane owns fixed (batch,col) cells
#pragma unroll
        for (int r = 0; r < 8; ++r) {
            int mrow = r + half * 8;
            int b = bb0 + mrow;
            if (r == 0 && t + 1 < Ldim) {  // pull next step's gate lines toward L2/L0
                size_t gb = ((size_t)b * Ldim + (t + 1)) * (size_t)G4 + col;
                __builtin_prefetch(xg + gb, 0, 0);
            }
            float iv = sigmoidf_(acc[0][r]);
            float fv = sigmoidf_(acc[1][r]);
            float gv = tanhfast_(acc[2][r]);
            float ov = sigmoidf_(acc[3][r]);
            float cn = fv * c_reg[r] + iv * gv;
            float hn = ov * tanhfast_(cn);
            float mv = mask[b * Ldim + t];
            float ho = mv * hn + (1.0f - mv) * h_reg[r];
            float co = mv * cn + (1.0f - mv) * c_reg[r];
            h_reg[r] = ho;
            c_reg[r] = co;
            hbuf[mrow * Hdim + col] = (_Float16)ho;
            out[((size_t)b * Ldim + t) * Hdim + col] = ho;
        }
        __syncthreads();   // hbuf updated before next step's A loads
    }
}

// ---------------- launch ----------------
extern "C" void kernel_launch(void* const* d_in, const int* in_sizes, int n_in,
                              void* d_out, int out_size, void* d_ws, size_t ws_size,
                              hipStream_t stream) {
    const float* x    = (const float*)d_in[0];
    const float* mask = (const float*)d_in[1];
    const float* w_ih = (const float*)d_in[2];
    const float* w_hh = (const float*)d_in[3];
    const float* b_ih = (const float*)d_in[4];
    const float* b_hh = (const float*)d_in[5];
    const float* h0   = (const float*)d_in[6];
    const float* c0   = (const float*)d_in[7];
    float* out = (float*)d_out;

    char* ws = (char*)d_ws;
    size_t off = 0;
    _Float16* xh    = (_Float16*)(ws + off); off += (size_t)Bdim * Ldim * Idim * 2;  // 64 MB
    _Float16* wihpk = (_Float16*)(ws + off); off += (size_t)G4 * Idim * 2;           // 2 MB
    _Float16* whhpk = (_Float16*)(ws + off); off += (size_t)G4 * Hdim * 2;           // 2 MB
    float*    xg    = (float*)(ws + off);    off += (size_t)Bdim * Ldim * G4 * 4;    // 512 MB

    {
        int n = Bdim * Ldim * Idim;
        k_cvt_f16<<<(n + 255) / 256, 256, 0, stream>>>(x, xh, n);
    }
    {
        int n = G4 * 512;
        k_pack_w<<<(n + 255) / 256, 256, 0, stream>>>(w_ih, wihpk);
        k_pack_w<<<(n + 255) / 256, 256, 0, stream>>>(w_hh, whhpk);
    }
    // (4096/2) M-tile pairs * 32 wave-columns / 8 waves-per-block = 8192 blocks
    k_xgates<<<8192, 256, 0, stream>>>(xh, wihpk, b_ih, b_hh, xg);
    // persistent scan: one block per 16-row batch tile, 32 waves
    k_lstm<<<4, 1024, 0, stream>>>(whhpk, xg, mask, h0, c0, out);
}
